// Sender_LSTM_19739669692729
// MI455X (gfx1250) — compile-verified, matrix operand order checked by reading, not run on hardware
//
#include <hip/hip_runtime.h>
#include <hip/hip_bf16.h>
#include <stdint.h>

// ---------------------------------------------------------------------------
// Types for CDNA5 WMMA (wave32) + TDM descriptors
// ---------------------------------------------------------------------------
typedef __attribute__((ext_vector_type(16))) __bf16 v16bf;
typedef __attribute__((ext_vector_type(8)))  __bf16 bf16x8;
typedef __attribute__((ext_vector_type(8)))  float  v8f;
typedef __attribute__((ext_vector_type(4)))  unsigned int u32x4;
typedef __attribute__((ext_vector_type(8)))  unsigned int u32x8;

#define CAT16(lo, hi) __builtin_shufflevector((lo), (hi), 0,1,2,3,4,5,6,7,8,9,10,11,12,13,14,15)

// ---------------------------------------------------------------------------
// Problem dimensions (fixed by the reference)
// ---------------------------------------------------------------------------
#define TT     32
#define BB     256
#define DD     512
#define HH     1024
#define VV     32000
#define H4     (4 * HH)
#define STEPS  15
#define OUTW   (2 * STEPS)   // 30 columns per output row (padded halves)

// ---------------------------------------------------------------------------
// Conversion kernels
// ---------------------------------------------------------------------------
__global__ __launch_bounds__(256) void f32_to_bf16_kernel(
    const float* __restrict__ in, __bf16* __restrict__ out, size_t n)
{
    size_t i = (size_t)blockIdx.x * blockDim.x + threadIdx.x;
    if (i < n) out[i] = (__bf16)in[i];
}

// in: [K][N] f32 row-major  ->  out: [N][K] bf16 row-major (transposed)
__global__ __launch_bounds__(256) void transpose_f32_to_bf16_kernel(
    const float* __restrict__ in, __bf16* __restrict__ out, int K, int N)
{
    size_t i = (size_t)blockIdx.x * blockDim.x + threadIdx.x;
    size_t total = (size_t)K * N;
    if (i < total) {
        int k = (int)(i / N);
        int n = (int)(i % N);
        out[(size_t)n * K + k] = (__bf16)in[i];
    }
}

// ---------------------------------------------------------------------------
// TDM: load a [rows x K] bf16 tile (row-major, contiguous rows of a larger
// [tensor_rows x K] tensor) into LDS at lds_off.  D# layout per
// CDNA5 ISA.  Tracked by TENSORcnt.
// ---------------------------------------------------------------------------
__device__ __forceinline__ void tdm_load_tile_to_lds(
    const void* gaddr, unsigned lds_off, int K, int rows, int tensor_rows)
{
    uint64_t ga = (uint64_t)gaddr;
    u32x4 g0;
    g0[0] = 1u;                                           // count=1, user desc
    g0[1] = lds_off;                                      // lds_addr (bytes)
    g0[2] = (unsigned)(ga & 0xFFFFFFFFu);                 // global_addr[31:0]
    g0[3] = (unsigned)((ga >> 32) & 0x01FFFFFFu)          // global_addr[56:32]
          | (2u << 30);                                   // type = 2 ("image")
    u32x8 g1;
    unsigned uK = (unsigned)K, uR = (unsigned)tensor_rows;
    g1[0] = (1u << 16);                                   // data_size=2B, mask=0
    g1[1] = (uK & 0xFFFFu) << 16;                         // tensor_dim0[15:0]
    g1[2] = (uK >> 16) | ((uR & 0xFFFFu) << 16);          // dim0[31:16] | dim1[15:0]
    g1[3] = ((uR >> 16) & 0xFFFFu) | ((uK & 0xFFFFu) << 16); // dim1[31:16] | tile_dim0
    g1[4] = (unsigned)rows & 0xFFFFu;                     // tile_dim1 (tile_dim2=0)
    g1[5] = uK;                                           // tensor_dim0_stride[31:0]
    g1[6] = 0u;                                           // stride0[47:32] | stride1[15:0]
    g1[7] = 0u;                                           // stride1[47:16]
    asm volatile("tensor_load_to_lds %0, %1" :: "s"(g0), "s"(g1) : "memory");
}

// ---------------------------------------------------------------------------
// bf16 WMMA GEMM:  C[M][N](f32) (+)= A[M][K](bf16,row) x Bt[N][K](bf16,row) + bias
// Block: 128 threads = 4 waves; block tile 32(M) x 256(N).
// Each wave owns a 32(M) x 64(N) strip: two A fragments (rows 0-15, 16-31 of
// the block tile) x 4 B fragments -> 8 WMMAs per K-step, with every B
// fragment reused twice (32 FLOP per L2 byte of weight traffic).
// A tile (32 x K) is DMA'd into LDS by the Tensor Data Mover once per block;
// A fragments come from LDS (ds_load_b128); B fragments are double-buffered
// global loads (2 K-steps in flight).
// Fragment layout per CDNA5 ISA (wave32, 16-bit A/B):
//   lane L: m/n = L%16, half = L/16; elems 0..7 -> K = half*8 + 0..7,
//   elems 8..15 -> K = 16 + half*8 + 0..7  (two contiguous 16B chunks).
// C/D: VGPR r holds (M = half*8 + r, N = lane%16).
// ---------------------------------------------------------------------------
__global__ __launch_bounds__(128) void gemm_bf16_wmma_kernel(
    const __bf16* __restrict__ A,    // [M][K]
    const __bf16* __restrict__ Bt,   // [N][K]
    float* __restrict__ C,           // [M][N]
    const float* __restrict__ bias,  // [N] or nullptr
    int M, int N, int K, int accumulate)
{
    extern __shared__ __bf16 smem[];             // [32][K] A tile

    const int lane = threadIdx.x & 31;
    const int wave = threadIdx.x >> 5;           // 0..3 -> N sub-tile
    const int row0 = blockIdx.y * 32;
    const int col0 = blockIdx.x * 256 + wave * 64;
    const int ml   = lane & 15;
    const int half = lane >> 4;

    // --- TDM: stage this block's A tile (rows blockIdx.y*32 .. +31) in LDS
    if (wave == 0) {
        tdm_load_tile_to_lds(A + (size_t)blockIdx.y * 32 * K, 0, K, 32, M);
    }

    // --- B fragment base pointers (global; weights are L2-resident)
    const __bf16* bbase[4];
#pragma unroll
    for (int j = 0; j < 4; ++j)
        bbase[j] = Bt + (size_t)(col0 + j * 16 + ml) * K + half * 8;

    // Prologue: prefetch B fragments for kk = 0 (overlaps the TDM transfer)
    bf16x8 lo0[4], hi0[4], lo1[4], hi1[4];
#pragma unroll
    for (int j = 0; j < 4; ++j) {
        lo0[j] = *(const bf16x8*)(bbase[j]);
        hi0[j] = *(const bf16x8*)(bbase[j] + 16);
    }

    v8f acc0[4], acc1[4];
#pragma unroll
    for (int j = 0; j < 4; ++j) {
        acc0[j] = (v8f){0.f,0.f,0.f,0.f,0.f,0.f,0.f,0.f};
        acc1[j] = (v8f){0.f,0.f,0.f,0.f,0.f,0.f,0.f,0.f};
    }

    // Wait for the A tile, then make it visible to all waves
    if (wave == 0) __builtin_amdgcn_s_wait_tensorcnt(0);
    __syncthreads();

    const __bf16* As0 = smem + (size_t)ml * K + half * 8;         // rows 0..15
    const __bf16* As1 = smem + (size_t)(16 + ml) * K + half * 8;  // rows 16..31

    // Main loop: 2 WMMA K-steps per iteration, double-buffered B fragments
    for (int kk = 0; kk < K; kk += 64) {
        // prefetch kk+32 into set 1
#pragma unroll
        for (int j = 0; j < 4; ++j) {
            lo1[j] = *(const bf16x8*)(bbase[j] + kk + 32);
            hi1[j] = *(const bf16x8*)(bbase[j] + kk + 48);
        }
        {
            v16bf a0 = CAT16(*(const bf16x8*)(As0 + kk), *(const bf16x8*)(As0 + kk + 16));
            v16bf a1 = CAT16(*(const bf16x8*)(As1 + kk), *(const bf16x8*)(As1 + kk + 16));
#pragma unroll
            for (int j = 0; j < 4; ++j) {
                v16bf b = CAT16(lo0[j], hi0[j]);
                acc0[j] = __builtin_amdgcn_wmma_f32_16x16x32_bf16(
                    false, a0, false, b, (short)0, acc0[j], false, false);
                acc1[j] = __builtin_amdgcn_wmma_f32_16x16x32_bf16(
                    false, a1, false, b, (short)0, acc1[j], false, false);
            }
        }
        // prefetch kk+64 into set 0
        if (kk + 64 < K) {
#pragma unroll
            for (int j = 0; j < 4; ++j) {
                lo0[j] = *(const bf16x8*)(bbase[j] + kk + 64);
                hi0[j] = *(const bf16x8*)(bbase[j] + kk + 80);
            }
        }
        {
            v16bf a0 = CAT16(*(const bf16x8*)(As0 + kk + 32), *(const bf16x8*)(As0 + kk + 48));
            v16bf a1 = CAT16(*(const bf16x8*)(As1 + kk + 32), *(const bf16x8*)(As1 + kk + 48));
#pragma unroll
            for (int j = 0; j < 4; ++j) {
                v16bf b = CAT16(lo1[j], hi1[j]);
                acc0[j] = __builtin_amdgcn_wmma_f32_16x16x32_bf16(
                    false, a0, false, b, (short)0, acc0[j], false, false);
                acc1[j] = __builtin_amdgcn_wmma_f32_16x16x32_bf16(
                    false, a1, false, b, (short)0, acc1[j], false, false);
            }
        }
    }

    // Epilogue: bias add, optional accumulate, strided f32 stores
#pragma unroll
    for (int j = 0; j < 4; ++j) {
        int n = col0 + j * 16 + ml;
        float bv = bias ? bias[n] : 0.0f;
#pragma unroll
        for (int r = 0; r < 8; ++r) {
            size_t idx0 = (size_t)(row0 + half * 8 + r) * N + n;
            float v0 = acc0[j][r] + bv;
            if (accumulate) v0 += C[idx0];
            C[idx0] = v0;
            size_t idx1 = (size_t)(row0 + 16 + half * 8 + r) * N + n;
            float v1 = acc1[j][r] + bv;
            if (accumulate) v1 += C[idx1];
            C[idx1] = v1;
        }
    }
}

// ---------------------------------------------------------------------------
// LSTM gate kernel (Keras order i,f,g,o; linear candidate; h = sigmoid(o)*c)
// ---------------------------------------------------------------------------
__device__ __forceinline__ float sigmoidf(float x) {
    return 1.0f / (1.0f + __expf(-x));
}

__global__ __launch_bounds__(256) void lstm_gates_kernel(
    const float* __restrict__ z,   // [B][4H]
    float* __restrict__ h,         // [B][H]
    float* __restrict__ c,         // [B][H]
    __bf16* __restrict__ h_bf)     // [B][H]
{
    int idx = blockIdx.x * blockDim.x + threadIdx.x;
    if (idx >= BB * HH) return;
    int b = idx / HH;
    int j = idx % HH;
    const float* zr = z + (size_t)b * H4;
    float ig = sigmoidf(zr[j]);
    float fg = sigmoidf(zr[HH + j]);
    float gg = zr[2 * HH + j];            // activation=None -> linear candidate
    float og = sigmoidf(zr[3 * HH + j]);
    float cn = fg * c[idx] + ig * gg;
    float hn = og * cn;                   // h = o * c (no tanh)
    c[idx] = cn;
    h[idx] = hn;
    h_bf[idx] = (__bf16)hn;
}

// ---------------------------------------------------------------------------
// log-softmax over V per row, in place; also entropy per row.
// One 256-thread block per row; LDS tree reductions.
// ---------------------------------------------------------------------------
__global__ __launch_bounds__(256) void logsoftmax_entropy_kernel(
    float* __restrict__ logits,   // [B][V], overwritten with logp
    float* __restrict__ ent)      // [B]
{
    __shared__ float red[256];
    const int b = blockIdx.x;
    float* row = logits + (size_t)b * VV;
    const int t = threadIdx.x;

    float m = -3.4e38f;
    for (int v = t; v < VV; v += 256) m = fmaxf(m, row[v]);
    red[t] = m; __syncthreads();
    for (int s = 128; s > 0; s >>= 1) {
        if (t < s) red[t] = fmaxf(red[t], red[t + s]);
        __syncthreads();
    }
    m = red[0]; __syncthreads();

    float sum = 0.f;
    for (int v = t; v < VV; v += 256) sum += __expf(row[v] - m);
    red[t] = sum; __syncthreads();
    for (int s = 128; s > 0; s >>= 1) {
        if (t < s) red[t] += red[t + s];
        __syncthreads();
    }
    float lse = m + __logf(red[0]); __syncthreads();

    float e = 0.f;
    for (int v = t; v < VV; v += 256) {
        float lp = row[v] - lse;
        row[v] = lp;
        e += __expf(lp) * lp;
    }
    red[t] = e; __syncthreads();
    for (int s = 128; s > 0; s >>= 1) {
        if (t < s) red[t] += red[t + s];
        __syncthreads();
    }
    if (t == 0) ent[b] = -red[0];
}

// ---------------------------------------------------------------------------
// Categorical sampling via inverse CDF + output packing.
// d_out layout (flat f32, reference return order):
//   message[B][30], logits[B][30], entropy[B][30], h[B][H]
// ---------------------------------------------------------------------------
__device__ __forceinline__ float uniform01(uint64_t x) {
    x += 0x9E3779B97F4A7C15ull;
    x = (x ^ (x >> 30)) * 0xBF58476D1CE4E5B9ull;
    x = (x ^ (x >> 27)) * 0x94D049BB133111EBull;
    x =  x ^ (x >> 31);
    return ((float)((x >> 40) & 0xFFFFFF) + 0.5f) * (1.0f / 16777216.0f);
}

__global__ __launch_bounds__(256) void sample_pack_kernel(
    const float* __restrict__ logp,  // [B][V]
    const float* __restrict__ ent,   // [B]
    float* __restrict__ out)
{
    int idx = blockIdx.x * blockDim.x + threadIdx.x;
    if (idx >= BB * OUTW) return;
    int b = idx / OUTW;
    int s = idx % OUTW;
    float* msg  = out;
    float* lgt  = out + (size_t)BB * OUTW;
    float* entr = out + (size_t)2 * BB * OUTW;
    size_t o = (size_t)b * OUTW + s;

    if (s < STEPS) {
        float u = uniform01(((uint64_t)s * BB + (uint64_t)b) ^ 0x2Aull /*seed 42*/);
        const float* row = logp + (size_t)b * VV;
        float acc = 0.f;
        int sel = VV - 1;
        for (int v = 0; v < VV; ++v) {
            acc += __expf(row[v]);
            if (acc >= u) { sel = v; break; }
        }
        msg[o]  = (float)sel;
        lgt[o]  = row[sel];
        entr[o] = ent[b];
    } else {
        msg[o]  = 0.f;
        lgt[o]  = 0.f;
        entr[o] = 0.f;
    }
}

__global__ __launch_bounds__(256) void copy_h_kernel(
    const float* __restrict__ h, float* __restrict__ out)
{
    int i = blockIdx.x * blockDim.x + threadIdx.x;
    if (i < BB * HH) out[(size_t)3 * BB * OUTW + i] = h[i];
}

// ---------------------------------------------------------------------------
// Host orchestration
// ---------------------------------------------------------------------------
extern "C" void kernel_launch(void* const* d_in, const int* in_sizes, int n_in,
                              void* d_out, int out_size, void* d_ws, size_t ws_size,
                              hipStream_t stream)
{
    (void)in_sizes; (void)n_in; (void)out_size; (void)ws_size;

    const float* x  = (const float*)d_in[0];  // [T][B][D]
    const float* Wi = (const float*)d_in[1];  // [D][4H]
    const float* Wh = (const float*)d_in[2];  // [H][4H]
    const float* bi = (const float*)d_in[3];  // [4H]
    const float* Wo = (const float*)d_in[4];  // [H][V]
    const float* bo = (const float*)d_in[5];  // [V]
    float* out = (float*)d_out;

    // Workspace layout (all offsets multiples of 256 bytes)
    char* ws = (char*)d_ws;
    size_t off = 0;
    __bf16* Wi_t = (__bf16*)(ws + off); off += (size_t)H4 * DD * 2;      //  4 MB  [4H][D]
    __bf16* Wh_t = (__bf16*)(ws + off); off += (size_t)H4 * HH * 2;      //  8 MB  [4H][H]
    __bf16* Wo_t = (__bf16*)(ws + off); off += (size_t)VV * HH * 2;      // 64 MB  [V][H]
    __bf16* x_bf = (__bf16*)(ws + off); off += (size_t)TT * BB * DD * 2; //  8 MB
    float*  z    = (float*) (ws + off); off += (size_t)BB * H4 * 4;      //  4 MB
    float*  h    = (float*) (ws + off); off += (size_t)BB * HH * 4;      //  1 MB
    float*  c    = (float*) (ws + off); off += (size_t)BB * HH * 4;      //  1 MB
    __bf16* h_bf = (__bf16*)(ws + off); off += (size_t)BB * HH * 2;      //  0.5 MB
    float*  lgts = (float*) (ws + off); off += (size_t)BB * VV * 4;      // 32.8 MB
    float*  ent  = (float*) (ws + off); off += (size_t)BB * 4;

    // 1) Precision conversion: weights -> bf16 transposed, x -> bf16
    {
        size_t n = (size_t)DD * H4;
        transpose_f32_to_bf16_kernel<<<(n + 255) / 256, 256, 0, stream>>>(Wi, Wi_t, DD, H4);
        n = (size_t)HH * H4;
        transpose_f32_to_bf16_kernel<<<(n + 255) / 256, 256, 0, stream>>>(Wh, Wh_t, HH, H4);
        n = (size_t)HH * VV;
        transpose_f32_to_bf16_kernel<<<(n + 255) / 256, 256, 0, stream>>>(Wo, Wo_t, HH, VV);
        n = (size_t)TT * BB * DD;
        f32_to_bf16_kernel<<<(n + 255) / 256, 256, 0, stream>>>(x, x_bf, n);
    }

    // 2) Zero LSTM state (graph-capture-safe async memsets)
    hipMemsetAsync(h,    0, (size_t)BB * HH * 4, stream);
    hipMemsetAsync(c,    0, (size_t)BB * HH * 4, stream);
    hipMemsetAsync(h_bf, 0, (size_t)BB * HH * 2, stream);

    // 3) Sequential LSTM: 32 steps of (z = x_t@Wi + b; z += h@Wh; gates)
    dim3 gemm_blk(128);                          // 4 waves per block
    dim3 gemm_grid_gate(H4 / 256, BB / 32);      // (16, 8)
    const size_t lds_xi = (size_t)32 * DD * 2;   // 32 KB A tile (K=512)
    const size_t lds_hh = (size_t)32 * HH * 2;   // 64 KB A tile (K=1024)
    for (int t = 0; t < TT; ++t) {
        const __bf16* xt = x_bf + (size_t)t * BB * DD;
        gemm_bf16_wmma_kernel<<<gemm_grid_gate, gemm_blk, lds_xi, stream>>>(
            xt, Wi_t, z, bi, BB, H4, DD, 0);
        gemm_bf16_wmma_kernel<<<gemm_grid_gate, gemm_blk, lds_hh, stream>>>(
            h_bf, Wh_t, z, nullptr, BB, H4, HH, 1);
        lstm_gates_kernel<<<(BB * HH) / 256, 256, 0, stream>>>(z, h, c, h_bf);
    }

    // 4) Output projection: logits = h @ Wo + bo   [256 x 32000, K=1024]
    dim3 gemm_grid_out(VV / 256, BB / 32);       // (125, 8)
    gemm_bf16_wmma_kernel<<<gemm_grid_out, gemm_blk, lds_hh, stream>>>(
        h_bf, Wo_t, lgts, bo, BB, VV, HH, 0);

    // 5) log-softmax + entropy (in place on lgts)
    logsoftmax_entropy_kernel<<<BB, 256, 0, stream>>>(lgts, ent);

    // 6) Sample + pack message/logits/entropy (with zero padding), copy h
    sample_pack_kernel<<<(BB * OUTW + 255) / 256, 256, 0, stream>>>(lgts, ent, out);
    copy_h_kernel<<<(BB * HH) / 256, 256, 0, stream>>>(h, out);
}
